// RobertaCRF_21552145891364
// MI455X (gfx1250) — matrix operand order repaired and checked by reading, not compiled, not run on hardware
//
#include <hip/hip_runtime.h>
#include <math.h>

#define Bsz 8
#define Sln 512
#define Hd  768
#define NHd 12
#define DHd 64
#define FFd 3072
#define NLy 12
#define NLb 10
#define Mtot (Bsz*Sln)

typedef unsigned short u16;
typedef __attribute__((ext_vector_type(16))) __bf16 bf16x16;
typedef __attribute__((ext_vector_type(8)))  float  floatx8;

__device__ __forceinline__ u16 f2bf(float f) {
  unsigned u = __float_as_uint(f);
  u += 0x7FFFu + ((u >> 16) & 1u);   // round-to-nearest-even
  return (u16)(u >> 16);
}

// Low 32 bits of a generic shared-memory pointer are the LDS byte offset
// (flat LDS aperture: addr[63:32]=aperture, addr[31:0]=LDS offset).
__device__ __forceinline__ unsigned lds_off32(const void* p) {
  return (unsigned)(unsigned long long)p;
}

union FragBF { uint4 q[2]; bf16x16 v; };

// ---------------- bf16 conversion ----------------
__global__ __launch_bounds__(256) void convert_bf16_kernel(
    const float* __restrict__ in, u16* __restrict__ out, int n) {
  int i = blockIdx.x * 256 + threadIdx.x;
  if (i < n) out[i] = f2bf(in[i]);
}

// ---------------- block reduce helpers (256 threads) ----------------
__device__ __forceinline__ float blk_sum256(float v, float* red) {
  int tid = threadIdx.x;
  red[tid] = v; __syncthreads();
  for (int st = 128; st > 0; st >>= 1) { if (tid < st) red[tid] += red[tid + st]; __syncthreads(); }
  float r = red[0]; __syncthreads(); return r;
}
__device__ __forceinline__ float blk_max256(float v, float* red) {
  int tid = threadIdx.x;
  red[tid] = v; __syncthreads();
  for (int st = 128; st > 0; st >>= 1) { if (tid < st) red[tid] = fmaxf(red[tid], red[tid + st]); __syncthreads(); }
  float r = red[0]; __syncthreads(); return r;
}

// ---------------- embedding + LN (writes fp32 x and bf16 xb) ----------------
__global__ __launch_bounds__(256) void embed_ln_kernel(
    const int* __restrict__ ids, const int* __restrict__ mask,
    const float* __restrict__ wemb, const float* __restrict__ pemb,
    const float* __restrict__ g, const float* __restrict__ bt,
    float* __restrict__ x, u16* __restrict__ xb)
{
  __shared__ float buf[Hd];
  __shared__ float red[256];
  __shared__ int  redi[256];
  int tok = blockIdx.x;
  int b = tok / Sln, s = tok - b * Sln;
  int tid = threadIdx.x;

  int part = 0;
  for (int j = tid; j <= s; j += 256) part += mask[b * Sln + j];
  redi[tid] = part; __syncthreads();
  for (int st = 128; st > 0; st >>= 1) { if (tid < st) redi[tid] += redi[tid + st]; __syncthreads(); }
  int pos = redi[0] * mask[b * Sln + s] + 1;
  __syncthreads();

  int id = ids[tok];
  float s1 = 0.f, s2 = 0.f;
  for (int h = tid; h < Hd; h += 256) {
    float e = wemb[(long long)id * Hd + h] + pemb[(long long)pos * Hd + h];
    buf[h] = e; s1 += e; s2 += e * e;
  }
  float mean = blk_sum256(s1, red) * (1.0f / Hd);
  float var  = blk_sum256(s2, red) * (1.0f / Hd) - mean * mean;
  float rstd = rsqrtf(var + 1e-5f);
  for (int h = tid; h < Hd; h += 256) {
    float o = (buf[h] - mean) * rstd * g[h] + bt[h];
    x[(long long)tok * Hd + h] = o;
    xb[(long long)tok * Hd + h] = f2bf(o);
  }
}

// ---------------- residual add + LN (in-place x; writes bf16 xb) ----------------
__global__ __launch_bounds__(256) void add_ln_kernel(
    float* __restrict__ x, const float* __restrict__ t,
    const float* __restrict__ g, const float* __restrict__ bt,
    u16* __restrict__ xb)
{
  __shared__ float buf[Hd];
  __shared__ float red[256];
  int tok = blockIdx.x, tid = threadIdx.x;
  float s1 = 0.f, s2 = 0.f;
  for (int h = tid; h < Hd; h += 256) {
    float v = x[(long long)tok * Hd + h] + t[(long long)tok * Hd + h];
    buf[h] = v; s1 += v; s2 += v * v;
  }
  float mean = blk_sum256(s1, red) * (1.0f / Hd);
  float var  = blk_sum256(s2, red) * (1.0f / Hd) - mean * mean;
  float rstd = rsqrtf(var + 1e-5f);
  for (int h = tid; h < Hd; h += 256) {
    float o = (buf[h] - mean) * rstd * g[h] + bt[h];
    x[(long long)tok * Hd + h] = o;
    xb[(long long)tok * Hd + h] = f2bf(o);
  }
}

// ---------------- WMMA GEMM: C = A(bf16,MxK) * B(bf16,KxN) [+bias] [gelu] ----------------
// Macro tile 128x128, 8 waves (4x2), wave tile 32x64 (2x4 subtiles of 16x16), K chunk 32.
// Double-buffered LDS; tiles staged with GLOBAL_LOAD_ASYNC_TO_LDS_B128 and the next
// tile's copies overlapped with the current tile's WMMAs (in-order ASYNCcnt waits).
__global__ __launch_bounds__(256) void wmma_gemm_kernel(
    const u16* __restrict__ A, const u16* __restrict__ B, const float* __restrict__ bias,
    void* __restrict__ Cout, int outBf16,
    int M, int N, int K, int lda, int ldb, int ldc,
    long long sA, long long sB, long long sC, int transB, int doGelu)
{
  __shared__ __align__(16) u16 As[2][128][48];   // 2 x 12 KB
  __shared__ __align__(16) u16 Bs[2][128][48];   // 2 x 12 KB, transposed: Bs[b][n][k]

  const int tid  = threadIdx.x;
  const int lane = tid & 31;
  const int w    = tid >> 5;
  const int waveM = w >> 1;          // 0..3
  const int waveN = w & 1;           // 0..1
  const int mrow  = lane & 15;       // row within 16x16 tile (also N-col for B/C)
  const int khalf = lane >> 4;       // 0/1 lane-group

  const long long z = blockIdx.z;
  const u16* Ap = A + z * sA;
  const u16* Bp = B + z * sB;
  const int bm = blockIdx.y * 128;
  const int bn = blockIdx.x * 128;

  // Tail tiles: pre-zero both LDS buffers once; guarded lanes never overwrite.
  if (bm + 128 > M || bn + 128 > N) {
    u16* af  = &As[0][0][0];
    u16* bfp = &Bs[0][0][0];
    for (int i = tid; i < 2 * 128 * 48; i += 256) { af[i] = 0; bfp[i] = 0; }
  }
  __syncthreads();

  floatx8 c[2][4];
  const floatx8 zf = {0.f,0.f,0.f,0.f,0.f,0.f,0.f,0.f};
#pragma unroll
  for (int i = 0; i < 2; ++i)
#pragma unroll
    for (int j = 0; j < 4; ++j) c[i][j] = zf;

  // Per-thread staging coordinates (loop-invariant)
  const int rowA  = tid >> 1;
  const int halfA = tid & 1;
  const int gmA   = bm + rowA;
  const int nlB   = tid & 127;
  const int khB   = tid >> 7;        // 0/1 -> k sub-range of 16
  const int gnB   = bn + nlB;

  // Stage one 128x32 A tile + 32x128 B tile (transposed) into buffer nb.
  auto stage = [&](int k0, int nb) {
    if (gmA < M) {    // 32B run, contiguous in global and LDS -> async copy
      const u16* gp = Ap + (long long)gmA * lda + k0 + halfA * 16;
      unsigned la = lds_off32(&As[nb][rowA][halfA * 16]);
      asm volatile("global_load_async_to_lds_b128 %0, %1, off"
                   :: "v"(la), "v"(gp) : "memory");
      asm volatile("global_load_async_to_lds_b128 %0, %1, off offset:16"
                   :: "v"(la), "v"(gp) : "memory");
      if (k0 + 32 < K)
        __builtin_prefetch((const void*)(gp + 32), 0, 1);
    }
    if (transB) {     // B[k][n] = Bp[n*ldb + k]: k-contiguous -> async copy
      if (gnB < N) {
        const u16* gp = Bp + (long long)gnB * ldb + k0 + khB * 16;
        unsigned lb = lds_off32(&Bs[nb][nlB][khB * 16]);
        asm volatile("global_load_async_to_lds_b128 %0, %1, off"
                     :: "v"(lb), "v"(gp) : "memory");
        asm volatile("global_load_async_to_lds_b128 %0, %1, off offset:16"
                     :: "v"(lb), "v"(gp) : "memory");
      }
    } else {          // B[k][n] = Bp[k*ldb + n]: transpose during staging
      if (gnB < N) {
#pragma unroll
        for (int kk = khB * 16; kk < khB * 16 + 16; ++kk)
          Bs[nb][nlB][kk] = Bp[(long long)(k0 + kk) * ldb + gnB];
      }
    }
  };

  stage(0, 0);
  int kb = 0;
  for (int k0 = 0; k0 < K; k0 += 32, kb ^= 1) {
    const bool more = (k0 + 32 < K);
    if (more) stage(k0 + 32, kb ^ 1);   // overlap next tile's copies with this tile's math

    // Asyncs complete in order per wave: waiting <= (count just issued for the next
    // tile) guarantees the current tile's copies have landed in LDS.
    if (more) {
      if (transB) asm volatile("s_wait_asynccnt 4" ::: "memory");
      else        asm volatile("s_wait_asynccnt 2" ::: "memory");
    } else {
      asm volatile("s_wait_asynccnt 0" ::: "memory");
    }
    __syncthreads();

    // A frag: lane(l) holds row l&15; halfs 0..7 = K[(l>>4)*8 ..], 8..15 = K[16+(l>>4)*8 ..]
    FragBF a[2];
#pragma unroll
    for (int ti = 0; ti < 2; ++ti) {
      int r = waveM * 32 + ti * 16 + mrow;
      a[ti].q[0] = *(const uint4*)&As[kb][r][khalf * 8];
      a[ti].q[1] = *(const uint4*)&As[kb][r][16 + khalf * 8];
    }
  // B frag: lane(l) holds col l&15; halfs 0..7 = K[(l>>4)*8 ..], 8..15 = K[16+(l>>4)*8 ..]
    FragBF bf[4];
#pragma unroll
    for (int tj = 0; tj < 4; ++tj) {
      int n = waveN * 64 + tj * 16 + mrow;
      bf[tj].q[0] = *(const uint4*)&Bs[kb][n][khalf * 8];
      bf[tj].q[1] = *(const uint4*)&Bs[kb][n][16 + khalf * 8];
    }
#pragma unroll
    for (int ti = 0; ti < 2; ++ti)
#pragma unroll
      for (int tj = 0; tj < 4; ++tj)
        c[ti][tj] = __builtin_amdgcn_wmma_f32_16x16x32_bf16(
            false, a[ti].v, false, bf[tj].v, (short)0, c[ti][tj], false, false);
    __syncthreads();   // all waves done reading buffer kb before it is re-staged
  }

  // store: C lane(l): N = l&15, M = 8*(l>>4) + r
#pragma unroll
  for (int ti = 0; ti < 2; ++ti) {
#pragma unroll
    for (int tj = 0; tj < 4; ++tj) {
      int gn = bn + waveN * 64 + tj * 16 + mrow;
      if (gn < N) {
        float bv = bias ? bias[gn] : 0.f;
#pragma unroll
        for (int r = 0; r < 8; ++r) {
          int gm = bm + waveM * 32 + ti * 16 + khalf * 8 + r;
          if (gm < M) {
            float v = c[ti][tj][r] + bv;
            if (doGelu) v = 0.5f * v * (1.0f + erff(v * 0.70710678118f));
            long long idx = z * sC + (long long)gm * ldc + gn;
            if (outBf16) ((u16*)Cout)[idx] = f2bf(v);
            else         ((float*)Cout)[idx] = v;
          }
        }
      }
    }
  }
}

// ---------------- softmax over 512 keys, writes bf16 probs ----------------
__global__ __launch_bounds__(256) void softmax_bf16_kernel(
    const float* __restrict__ sc, u16* __restrict__ pr,
    const int* __restrict__ maskb, float scale)
{
  __shared__ float red[256];
  int z = blockIdx.y, row = blockIdx.x, tid = threadIdx.x;
  const float* rp = sc + ((long long)z * Sln + row) * Sln;
  float v0 = rp[tid]       * scale + (1.0f - (float)maskb[tid])       * (-1e9f);
  float v1 = rp[tid + 256] * scale + (1.0f - (float)maskb[tid + 256]) * (-1e9f);
  float m  = blk_max256(fmaxf(v0, v1), red);
  float e0 = expf(v0 - m), e1 = expf(v1 - m);
  float ssum = blk_sum256(e0 + e1, red);
  float inv = 1.0f / ssum;
  u16* op = pr + ((long long)z * Sln + row) * Sln;
  op[tid]       = f2bf(e0 * inv);
  op[tid + 256] = f2bf(e1 * inv);
}

// ---------------- emissions head: [4096,768] x [768,10] + b ----------------
__global__ __launch_bounds__(128) void emissions_kernel(
    const float* __restrict__ x, const float* __restrict__ hw,
    const float* __restrict__ hb, float* __restrict__ em)
{
  __shared__ float red[128];
  int tok = blockIdx.x, tid = threadIdx.x;
  float acc[NLb];
#pragma unroll
  for (int l = 0; l < NLb; ++l) acc[l] = 0.f;
  for (int h = tid; h < Hd; h += 128) {
    float xv = x[(long long)tok * Hd + h];
#pragma unroll
    for (int l = 0; l < NLb; ++l) acc[l] += xv * hw[h * NLb + l];
  }
  for (int l = 0; l < NLb; ++l) {
    red[tid] = acc[l]; __syncthreads();
    for (int st = 64; st > 0; st >>= 1) { if (tid < st) red[tid] += red[tid + st]; __syncthreads(); }
    if (tid == 0) em[(long long)tok * NLb + l] = red[0] + hb[l];
    __syncthreads();
  }
}

// ---------------- CRF forward (one block per batch) ----------------
__global__ void crf_kernel(
    const float* __restrict__ em, const int* __restrict__ labels, const int* __restrict__ mask,
    const float* __restrict__ startp, const float* __restrict__ endp,
    const float* __restrict__ trans, float* __restrict__ nll)
{
  __shared__ float alpha[NLb];
  int b = blockIdx.x, j = threadIdx.x;
  const float* e  = em + (long long)b * Sln * NLb;
  const int*   lb = labels + b * Sln;
  const int*   mk = mask + b * Sln;

  if (j < NLb) alpha[j] = startp[j] + e[j];
  float num = 0.f;
  int prev = lb[0];
  if (j == 0) num = startp[prev] + e[prev];
  __syncthreads();

  for (int t = 1; t < Sln; ++t) {
    float mt = (float)mk[t];
    float nxt = 0.f;
    if (j < NLb) {
      float mx = -1e30f;
      for (int i = 0; i < NLb; ++i) mx = fmaxf(mx, alpha[i] + trans[i * NLb + j]);
      float ss = 0.f;
      for (int i = 0; i < NLb; ++i) ss += expf(alpha[i] + trans[i * NLb + j] - mx);
      nxt = mx + logf(ss) + e[t * NLb + j];
    }
    __syncthreads();
    if (j < NLb && mt > 0.f) alpha[j] = nxt;
    if (j == 0) {
      int tt = lb[t];
      num += mt * (trans[prev * NLb + tt] + e[t * NLb + tt]);
      prev = tt;
    }
    __syncthreads();
  }
  if (j == 0) {
    int cnt = 0;
    for (int t = 0; t < Sln; ++t) cnt += mk[t];
    num += endp[lb[cnt - 1]];
    float mx = -1e30f;
    for (int i = 0; i < NLb; ++i) mx = fmaxf(mx, alpha[i] + endp[i]);
    float ss = 0.f;
    for (int i = 0; i < NLb; ++i) ss += expf(alpha[i] + endp[i] - mx);
    float logZ = mx + logf(ss);
    nll[b] = logZ - num;
  }
}

__global__ void crf_mean_kernel(const float* __restrict__ nll, float* __restrict__ out) {
  if (threadIdx.x == 0 && blockIdx.x == 0) {
    float s = 0.f;
    for (int i = 0; i < Bsz; ++i) s += nll[i];
    out[0] = s / (float)Bsz;
  }
}

// ---------------- host orchestration ----------------
extern "C" void kernel_launch(void* const* d_in, const int* in_sizes, int n_in,
                              void* d_out, int out_size, void* d_ws, size_t ws_size,
                              hipStream_t stream) {
  (void)in_sizes; (void)n_in; (void)out_size; (void)ws_size;
  const int*   ids      = (const int*)d_in[0];
  const int*   amask    = (const int*)d_in[1];
  const int*   labels   = (const int*)d_in[2];
  const float* word_emb = (const float*)d_in[3];
  const float* pos_emb  = (const float*)d_in[4];
  const float* emb_g    = (const float*)d_in[5];
  const float* emb_bt   = (const float*)d_in[6];
  const float* wq = (const float*)d_in[7];  const float* bq = (const float*)d_in[8];
  const float* wk = (const float*)d_in[9];  const float* bk = (const float*)d_in[10];
  const float* wv = (const float*)d_in[11]; const float* bv = (const float*)d_in[12];
  const float* wo = (const float*)d_in[13]; const float* bo = (const float*)d_in[14];
  const float* g1 = (const float*)d_in[15]; const float* c1 = (const float*)d_in[16];
  const float* w1 = (const float*)d_in[17]; const float* b1 = (const float*)d_in[18];
  const float* w2 = (const float*)d_in[19]; const float* b2 = (const float*)d_in[20];
  const float* g2 = (const float*)d_in[21]; const float* c2 = (const float*)d_in[22];
  const float* head_w = (const float*)d_in[23]; const float* head_b = (const float*)d_in[24];
  const float* crf_s = (const float*)d_in[25]; const float* crf_e = (const float*)d_in[26];
  const float* crf_t = (const float*)d_in[27];

  size_t off = 0;
  auto alloc = [&](size_t bytes) -> void* {
    void* p = (char*)d_ws + off;
    off = (off + bytes + 255) & ~(size_t)255;
    return p;
  };

  float* x     = (float*)alloc((size_t)Mtot * Hd * 4);
  u16*   xb    = (u16*)  alloc((size_t)Mtot * Hd * 2);
  u16*   qb    = (u16*)  alloc((size_t)Mtot * Hd * 2);
  u16*   kb    = (u16*)  alloc((size_t)Mtot * Hd * 2);
  u16*   vb    = (u16*)  alloc((size_t)Mtot * Hd * 2);
  u16*   ctxb  = (u16*)  alloc((size_t)Mtot * Hd * 2);
  u16*   h1b   = (u16*)  alloc((size_t)Mtot * FFd * 2);
  float* tbuf  = (float*)alloc((size_t)Mtot * Hd * 4);
  float* scores= (float*)alloc((size_t)NHd * Sln * Sln * 4);
  u16*   probs = (u16*)  alloc((size_t)NHd * Sln * Sln * 2);
  u16*   wqb   = (u16*)  alloc((size_t)Hd * Hd * 2);
  u16*   wkb   = (u16*)  alloc((size_t)Hd * Hd * 2);
  u16*   wvb   = (u16*)  alloc((size_t)Hd * Hd * 2);
  u16*   wob   = (u16*)  alloc((size_t)Hd * Hd * 2);
  u16*   w1b   = (u16*)  alloc((size_t)Hd * FFd * 2);
  u16*   w2b   = (u16*)  alloc((size_t)FFd * Hd * 2);
  float* em    = (float*)alloc((size_t)Mtot * NLb * 4);
  float* nll   = (float*)alloc(64);

  auto conv = [&](const float* src, u16* dst, int n) {
    convert_bf16_kernel<<<(n + 255) / 256, 256, 0, stream>>>(src, dst, n);
  };
  auto gemm = [&](const u16* A, const u16* Bm, const float* bias, void* C, int outBf16,
                  int M, int N, int K, int lda, int ldb, int ldc,
                  long long sA, long long sB, long long sC, int batch, int transB, int gelu) {
    dim3 g((N + 127) / 128, (M + 127) / 128, batch);
    wmma_gemm_kernel<<<g, 256, 0, stream>>>(A, Bm, bias, C, outBf16,
                                            M, N, K, lda, ldb, ldc, sA, sB, sC, transB, gelu);
  };

  embed_ln_kernel<<<Mtot, 256, 0, stream>>>(ids, amask, word_emb, pos_emb, emb_g, emb_bt, x, xb);

  for (int l = 0; l < NLy; ++l) {
    conv(wq + (size_t)l * Hd * Hd,  wqb, Hd * Hd);
    conv(wk + (size_t)l * Hd * Hd,  wkb, Hd * Hd);
    conv(wv + (size_t)l * Hd * Hd,  wvb, Hd * Hd);
    conv(wo + (size_t)l * Hd * Hd,  wob, Hd * Hd);
    conv(w1 + (size_t)l * Hd * FFd, w1b, Hd * FFd);
    conv(w2 + (size_t)l * FFd * Hd, w2b, FFd * Hd);

    // QKV projections (bf16 outputs, fused bias)
    gemm(xb, wqb, bq + (size_t)l * Hd, qb, 1, Mtot, Hd, Hd, Hd, Hd, Hd, 0, 0, 0, 1, 0, 0);
    gemm(xb, wkb, bk + (size_t)l * Hd, kb, 1, Mtot, Hd, Hd, Hd, Hd, Hd, 0, 0, 0, 1, 0, 0);
    gemm(xb, wvb, bv + (size_t)l * Hd, vb, 1, Mtot, Hd, Hd, Hd, Hd, Hd, 0, 0, 0, 1, 0, 0);

    for (int b = 0; b < Bsz; ++b) {
      // scores[h,q,k] = Q(512x64) * K^T(64x512), batched over heads
      gemm(qb + (size_t)b * Sln * Hd, kb + (size_t)b * Sln * Hd, nullptr, scores, 0,
           Sln, Sln, DHd, Hd, Hd, Sln, 64, 64, (long long)Sln * Sln, NHd, 1, 0);
      softmax_bf16_kernel<<<dim3(Sln, NHd), 256, 0, stream>>>(scores, probs, amask + b * Sln, 0.125f);
      // ctx[q,d] = P(512x512) * V(512x64), write back into [s, h*64+d] layout
      gemm(probs, vb + (size_t)b * Sln * Hd, nullptr, ctxb + (size_t)b * Sln * Hd, 1,
           Sln, DHd, Sln, Sln, Hd, Hd, (long long)Sln * Sln, 64, 64, NHd, 0, 0);
    }

    gemm(ctxb, wob, bo + (size_t)l * Hd, tbuf, 0, Mtot, Hd, Hd, Hd, Hd, Hd, 0, 0, 0, 1, 0, 0);
    add_ln_kernel<<<Mtot, 256, 0, stream>>>(x, tbuf, g1 + (size_t)l * Hd, c1 + (size_t)l * Hd, xb);

    gemm(xb, w1b, b1 + (size_t)l * FFd, h1b, 1, Mtot, FFd, Hd, Hd, FFd, FFd, 0, 0, 0, 1, 0, 1);
    gemm(h1b, w2b, b2 + (size_t)l * Hd, tbuf, 0, Mtot, Hd, FFd, FFd, Hd, Hd, 0, 0, 0, 1, 0, 0);
    add_ln_kernel<<<Mtot, 256, 0, stream>>>(x, tbuf, g2 + (size_t)l * Hd, c2 + (size_t)l * Hd, xb);
  }

  emissions_kernel<<<Mtot, 128, 0, stream>>>(x, head_w, head_b, em);
  crf_kernel<<<Bsz, 32, 0, stream>>>(em, labels, amask, crf_s, crf_e, crf_t, nll);
  crf_mean_kernel<<<1, 32, 0, stream>>>(nll, (float*)d_out);
}
